// ModulatedConv1d_3667902071005
// MI455X (gfx1250) — matrix operand order repaired and checked by reading, not compiled, not run on hardware
//
#include <hip/hip_runtime.h>
#include <hip/hip_bf16.h>
#include <stdint.h>

#define B_     16
#define CIN_   512
#define COUT_  512
#define KW_    3
#define SDIM_  512
#define L_     4096
#define NCHUNK (CIN_ / 32)   // 16 cin-chunks of 32 (WMMA K)

typedef __attribute__((ext_vector_type(16))) __bf16 v16bf;
typedef __attribute__((ext_vector_type(8)))  float  v8f;
typedef __attribute__((ext_vector_type(4)))  unsigned int v4u;
typedef __attribute__((ext_vector_type(8)))  int v8i;
typedef __attribute__((ext_vector_type(4)))  int v4i;

#if __has_builtin(__builtin_amdgcn_tensor_load_to_lds) && \
    __has_builtin(__builtin_amdgcn_s_wait_tensorcnt)
#define HAVE_TDM 1
#pragma message("CDNA5: TDM tensor_load_to_lds path ENABLED")
#else
#pragma message("CDNA5: TDM builtins missing -- manual staging fallback")
#endif

union Frag {
    v16bf v;
    uint4 q[2];
    unsigned short h[16];
};

static __device__ __forceinline__ unsigned short f2bf(float f) {
    unsigned int u = __float_as_uint(f);
    unsigned int r = u + 0x7FFFu + ((u >> 16) & 1u);   // round-to-nearest-even
    return (unsigned short)(r >> 16);
}

#ifdef HAVE_TDM
// ---------------------------------------------------------------------------
// Issue a 2D TDM tensor_load_to_lds of bf16 (data_size=2B) elements.
//   dim0: contiguous run of tile_d0 elements
//   dim1: tile_d1 rows, stride0 elements apart
// Descriptor packing per CDNA5 ISA §8.3/§8.4 (group0: count/lds/global/type,
// group1: data_size, tensor dims, tile dims, dim0 stride).
// ---------------------------------------------------------------------------
static __device__ __forceinline__ void tdm_load_bf16_2d(
        unsigned lds_addr, const void* gptr,
        unsigned tensor_d0, unsigned tensor_d1,
        unsigned tile_d0, unsigned tile_d1,
        unsigned long long stride0_elems) {
    unsigned long long ga = (unsigned long long)(uintptr_t)gptr;
    v4u g0;
    g0[0] = 1u;                                   // count=1, user load descriptor
    g0[1] = lds_addr;                             // LDS byte address
    g0[2] = (unsigned)ga;                         // global_addr[31:0]
    g0[3] = (unsigned)(ga >> 32) | (2u << 30);    // global_addr[56:32] | type=2
    v8i g1;
    g1[0] = (int)(1u << 16);                                          // data_size=2B
    g1[1] = (int)((tensor_d0 & 0xFFFFu) << 16);                       // td0[15:0]
    g1[2] = (int)((tensor_d0 >> 16) | ((tensor_d1 & 0xFFFFu) << 16)); // td0[31:16]|td1[15:0]
    g1[3] = (int)((tensor_d1 >> 16) | (tile_d0 << 16));               // td1[31:16]|tile_d0
    g1[4] = (int)(tile_d1 & 0xFFFFu);                                 // tile_d1 | tile_d2=0
    g1[5] = (int)(unsigned)(stride0_elems & 0xFFFFFFFFull);           // stride0[31:0]
    g1[6] = (int)(unsigned)(stride0_elems >> 32);                     // stride0[47:32]
    g1[7] = 0;
    v4i z4 = {0, 0, 0, 0};
#if __clang_major__ >= 23
    v8i z8 = {0, 0, 0, 0, 0, 0, 0, 0};
    __builtin_amdgcn_tensor_load_to_lds(g0, g1, z4, z4, z8, 0);
#else
    __builtin_amdgcn_tensor_load_to_lds(g0, g1, z4, z4, 0);
#endif
}
#endif

// ---------------------------------------------------------------------------
// Kernel 1: s[b][ci] = style[b] . mod_w[ci] + mod_b[ci]
// ---------------------------------------------------------------------------
__global__ void mod_style_kernel(const float* __restrict__ style,
                                 const float* __restrict__ mw,
                                 const float* __restrict__ mb,
                                 float* __restrict__ s) {
    int b  = blockIdx.x;
    int ci = threadIdx.x;                 // 512 threads
    const float* st = style + (size_t)b * SDIM_;
    const float* w  = mw    + (size_t)ci * SDIM_;
    float acc = 0.f;
#pragma unroll 8
    for (int j = 0; j < SDIM_; ++j) acc += st[j] * w[j];
    s[b * CIN_ + ci] = acc + mb[ci];
}

// ---------------------------------------------------------------------------
// Kernel 2: modulate + demodulate weights -> bf16, layout [b][cc][tap][co][ci32]
// ---------------------------------------------------------------------------
__global__ void mod_weight_kernel(const float* __restrict__ weight,  // [co][ci][k]
                                  const float* __restrict__ s,       // [b][ci]
                                  unsigned short* __restrict__ wb) {
    int blk = blockIdx.x;
    int b   = blk >> 9;
    int co  = blk & 511;
    const float scale = 0.0147313913f;   // 1/sqrt(512*3*3)

    __shared__ float red[256];
    __shared__ float sdemod;

    float part = 0.f;
    for (int idx = threadIdx.x; idx < CIN_ * KW_; idx += 256) {
        int ci = idx / 3, kk = idx - ci * 3;
        float wv = scale * weight[((size_t)co * CIN_ + ci) * KW_ + kk] * s[b * CIN_ + ci];
        part += wv * wv;
    }
    red[threadIdx.x] = part;
    __syncthreads();
    for (int off = 128; off > 0; off >>= 1) {
        if (threadIdx.x < off) red[threadIdx.x] += red[threadIdx.x + off];
        __syncthreads();
    }
    if (threadIdx.x == 0) sdemod = rsqrtf(red[0] + 1e-8f);
    __syncthreads();
    float dm = sdemod;

    for (int idx = threadIdx.x; idx < CIN_ * KW_; idx += 256) {
        int ci = idx / 3, kk = idx - ci * 3;
        float wv = scale * weight[((size_t)co * CIN_ + ci) * KW_ + kk] * s[b * CIN_ + ci] * dm;
        int cc  = ci >> 5;
        int cil = ci & 31;
        size_t o = ((((size_t)(b * NCHUNK + cc) * KW_ + kk) * COUT_ + co) << 5) + cil;
        wb[o] = f2bf(wv);
    }
}

// ---------------------------------------------------------------------------
// Kernel 3: convert input fp32 [b][ci][l] -> bf16 transposed [b][l][ci]
// ---------------------------------------------------------------------------
__global__ void convert_transpose_kernel(const float* __restrict__ in,
                                         unsigned short* __restrict__ it) {
    __shared__ float tile[32][33];
    int b   = blockIdx.z;
    int ciB = blockIdx.y * 32;
    int lB  = blockIdx.x * 32;
    int tx  = threadIdx.x;    // 32
    int ty  = threadIdx.y;    // 8
#pragma unroll
    for (int k = 0; k < 4; ++k) {
        int cil = ty + 8 * k;
        tile[cil][tx] = in[((size_t)b * CIN_ + ciB + cil) * L_ + lB + tx];
    }
    __syncthreads();
#pragma unroll
    for (int k = 0; k < 4; ++k) {
        int ll = ty + 8 * k;
        it[((size_t)b * L_ + lB + ll) * CIN_ + ciB + tx] = f2bf(tile[tx][ll]);
    }
}

// ---------------------------------------------------------------------------
// Kernel 4: conv-as-GEMM.  128 cout x 128 l per workgroup, 8 waves of 32x64.
// TDM double-buffered LDS staging: wave0 issues tensor_load_to_lds for the
// next cin-chunk while all waves run WMMA on the current one.
// ---------------------------------------------------------------------------
__global__ __launch_bounds__(256) void modconv_wmma_kernel(
        const unsigned short* __restrict__ wb,   // [b][cc][tap][co][32] bf16
        const unsigned short* __restrict__ it,   // [b][l][ci] bf16
        const float* __restrict__ bias,
        float* __restrict__ out) {               // [b][co][l] fp32
    __shared__ __align__(16) unsigned short lA[2][3][128][32];  // 48 KB
    __shared__ __align__(16) unsigned short lB[2][130][32];     // 16.6 KB

    int b      = blockIdx.z;
    int coBase = blockIdx.y * 128;
    int lBase  = blockIdx.x * 128;
    int tid    = threadIdx.x;
    int lane   = tid & 31;
    int wave   = tid >> 5;     // 0..7
    int wm     = wave & 3;     // cout sub-row (32 each)
    int wn     = wave >> 2;    // l sub-col (64 each)

    // halo clipping at sequence edges (TDM tile skips these rows; zero once)
    int hClip0 = (lBase == 0) ? 1 : 0;
    int hClip1 = (lBase + 128 >= L_) ? 1 : 0;
    int rows   = 130 - hClip0 - hClip1;

#ifdef HAVE_TDM
    auto issue = [&](int cc, int buf) {
        // A tile: 3 taps (rows) x 4096 contiguous bf16, row stride COUT_*32
        const unsigned short* ga =
            wb + ((size_t)((b * NCHUNK + cc) * 3) * COUT_ + coBase) * 32;
        tdm_load_bf16_2d((unsigned)(uintptr_t)&lA[buf][0][0][0], ga,
                         4096u, 3u, 4096u, 3u,
                         (unsigned long long)(COUT_ * 32));
        // B tile: `rows` haloed l-rows x 32 contiguous bf16, row stride CIN_
        const unsigned short* gb =
            it + (size_t)(b * L_ + lBase - 1 + hClip0) * CIN_ + cc * 32;
        tdm_load_bf16_2d((unsigned)(uintptr_t)&lB[buf][hClip0][0], gb,
                         32u, (unsigned)rows, 32u, (unsigned)rows,
                         (unsigned long long)CIN_);
    };
    if (wave == 0) issue(0, 0);     // prologue: chunk 0 into buffer 0
#endif

    // zero the halo rows TDM never writes (safe: issued loads skip them)
    if (hClip0 && tid < 8) ((uint4*)&lB[tid >> 2][0][0])[tid & 3]   = make_uint4(0, 0, 0, 0);
    if (hClip1 && tid < 8) ((uint4*)&lB[tid >> 2][129][0])[tid & 3] = make_uint4(0, 0, 0, 0);

    v8f acc[2][4];
#pragma unroll
    for (int i = 0; i < 2; ++i)
#pragma unroll
        for (int j = 0; j < 4; ++j)
#pragma unroll
            for (int e = 0; e < 8; ++e) acc[i][j][e] = 0.f;

    int m  = lane & 15;        // A row / B col / C col within 16x16 tile
    int kh = lane >> 4;        // which K-half this lane holds

    for (int cc = 0; cc < NCHUNK; ++cc) {
        int p = cc & 1;
#ifdef HAVE_TDM
        if (wave == 0) __builtin_amdgcn_s_wait_tensorcnt(0);
        __syncthreads();   // chunk cc in LDS; prev compute on buf p^1 finished
        if (wave == 0 && cc + 1 < NCHUNK) issue(cc + 1, p ^ 1);
#else
        __syncthreads();
        {   // manual staging fallback
            const uint4* wbq = (const uint4*)wb;
            const uint4* itq = (const uint4*)it;
            uint4* lAq = (uint4*)&lA[p][0][0][0];
            uint4* lBq = (uint4*)&lB[p][0][0];
            for (int i = tid; i < 1536; i += 256) {
                int tap = i >> 9, rem = i & 511;
                size_t src = ((size_t)((b * NCHUNK + cc) * 3 + tap) * COUT_ + coBase) * 4;
                lAq[i] = wbq[src + rem];
            }
            for (int i = tid; i < 520; i += 256) {
                int row = i >> 2, part = i & 3;
                int l = lBase - 1 + row;
                uint4 v = make_uint4(0, 0, 0, 0);
                if (l >= 0 && l < L_)
                    v = itq[(((size_t)b * L_ + l) << 6) + (cc << 2) + part];
                lBq[i] = v;
            }
        }
        __syncthreads();
#endif

#pragma unroll
        for (int tap = 0; tap < 3; ++tap) {
            Frag a[2], bm[4];
#pragma unroll
            for (int mt = 0; mt < 2; ++mt) {
                // A 16x32 bf16 layout: row m=lane&15, K chunks {kh*8.., 16+kh*8..}
                const unsigned short* pA = &lA[p][tap][wm * 32 + mt * 16 + m][kh * 8];
                a[mt].q[0] = *(const uint4*)pA;
                a[mt].q[1] = *(const uint4*)(pA + 16);
            }
#pragma unroll
            for (int nt = 0; nt < 4; ++nt) {
                // B 32x16 bf16 layout: col n=lane&15, K chunks {kh*8.., 16+kh*8..}
                const unsigned short* pB = &lB[p][wn * 64 + nt * 16 + m + tap][kh * 8];
                bm[nt].q[0] = *(const uint4*)pB;
                bm[nt].q[1] = *(const uint4*)(pB + 16);
            }
#pragma unroll
            for (int mt = 0; mt < 2; ++mt)
#pragma unroll
                for (int nt = 0; nt < 4; ++nt)
                    acc[mt][nt] = __builtin_amdgcn_wmma_f32_16x16x32_bf16(
                        false, a[mt].v, false, bm[nt].v,
                        (short)0, acc[mt][nt], false, false);
        }
    }

    // ---- epilogue: C layout VGPR r: lanes0-15 M=r, lanes16-31 M=r+8; N=lane&15
    int mhalf = kh * 8;
#pragma unroll
    for (int mt = 0; mt < 2; ++mt)
#pragma unroll
        for (int nt = 0; nt < 4; ++nt) {
#pragma unroll
            for (int r = 0; r < 8; ++r) {
                int co = coBase + wm * 32 + mt * 16 + mhalf + r;
                int l  = lBase + wn * 64 + nt * 16 + m;
                out[((size_t)b * COUT_ + co) * L_ + l] = acc[mt][nt][r] + bias[co];
            }
        }
}

// ---------------------------------------------------------------------------
extern "C" void kernel_launch(void* const* d_in, const int* in_sizes, int n_in,
                              void* d_out, int out_size, void* d_ws, size_t ws_size,
                              hipStream_t stream) {
    const float* input  = (const float*)d_in[0];   // (16,512,4096)
    const float* style  = (const float*)d_in[1];   // (16,512)
    const float* weight = (const float*)d_in[2];   // (1,512,512,3)
    const float* bias   = (const float*)d_in[3];   // (512,)
    const float* mod_w  = (const float*)d_in[4];   // (512,512)
    const float* mod_b  = (const float*)d_in[5];   // (512,)
    float* out = (float*)d_out;

    char* ws = (char*)d_ws;
    float* s_mod = (float*)ws;                                    // 32 KB
    unsigned short* wb = (unsigned short*)(ws + 64 * 1024);       // 25.2 MB
    size_t wb_bytes = (size_t)B_ * NCHUNK * KW_ * COUT_ * 32 * 2;
    unsigned short* it = (unsigned short*)(ws + 64 * 1024 + ((wb_bytes + 255) & ~(size_t)255)); // 67 MB

    mod_style_kernel<<<B_, 512, 0, stream>>>(style, mod_w, mod_b, s_mod);
    mod_weight_kernel<<<B_ * COUT_, 256, 0, stream>>>(weight, s_mod, wb);
    dim3 tgrid(L_ / 32, CIN_ / 32, B_);
    convert_transpose_kernel<<<tgrid, dim3(32, 8, 1), 0, stream>>>(input, it);
    dim3 cgrid(L_ / 128, COUT_ / 128, B_);
    modconv_wmma_kernel<<<cgrid, 256, 0, stream>>>(wb, it, bias, out);
}
